// PostProcessCocoGrounding_4157528342715
// MI455X (gfx1250) — matrix operand order repaired and checked by reading, not compiled, not run on hardware
//
#include <hip/hip_runtime.h>
#include <hip/hip_bf16.h>

typedef __attribute__((ext_vector_type(16))) _Float16 v16h;
typedef __attribute__((ext_vector_type(8)))  float    v8f;
typedef __attribute__((ext_vector_type(4)))  float    v4f;

#define B_BATCH   128
#define Q_DIM     900
#define T_DIM     256
#define C_DIM     91
#define C_PAD     96
#define NUM_SEL   300
#define CAND_CAP  2048
#define TOPK_THREADS 1024

#define KSTEPS    (T_DIM / 32)     // 8
#define NTILES    (C_PAD / 16)     // 6
// fragment-ready B: [kstep][ntile][lane][16 halves]
#define LDSB_ELEMS (KSTEPS * NTILES * 32 * 16)   // 24576 halves = 49152 B

// Fast sigmoid: v_exp_f32 + v_add + v_rcp_f32 (native), no IEEE divide chain.
// Denominator is in [1,2] -> rcp is well-conditioned; result feeds f16 WMMA.
__device__ __forceinline__ float sigmoidf_fast(float x) {
    return __builtin_amdgcn_rcpf(1.0f + __expf(-x));
}

// ---------------------------------------------------------------------------
// Kernel 1: prob[b,q,c] = sum_t sigmoid(logits[b,q,t]) * posmap[c,t]
// 128 threads = 4 waves; each wave computes a 16(M) x 96(N) strip, K = 256.
// B is staged once per block into a *fragment-ready* swizzled LDS layout so
// each V_WMMA_F32_16X16X32_F16 B-operand is one contiguous 32-byte LDS read
// (2x ds_load_b128). All 6 B fragments of a K-step are loaded before the 6
// WMMAs so the DS loads clause together instead of wait-per-WMMA.
// grid = (ceil(900/64)=15, 128)
// ---------------------------------------------------------------------------
__global__ __launch_bounds__(128)
void sigmoid_gemm_wmma(const float* __restrict__ logits,
                       const float* __restrict__ posmap,
                       float* __restrict__ prob)
{
    __shared__ __align__(32) _Float16 ldsB[LDSB_ELEMS];

    const int tid    = threadIdx.x;
    const int b      = blockIdx.y;
    const int qblock = blockIdx.x * 64;

    // ---- Stage positive_map^T into fragment-ready layout.
    // Element j of lane L for tile (kstep, n):
    //   K = kstep*32 + 16*(L>>4) + j,   C = n*16 + (L&15)
    for (int i = tid; i < LDSB_ELEMS; i += 128) {
        const int j     = i & 15;
        const int ln    = (i >> 4) & 31;
        const int n     = (i >> 9) % NTILES;
        const int kstep = i / (16 * 32 * NTILES);
        const int k     = kstep * 32 + ((ln >> 4) << 4) + j;
        const int c     = n * 16 + (ln & 15);
        const float v   = (c < C_DIM) ? posmap[c * T_DIM + k] : 0.0f;
        ldsB[i] = (_Float16)v;
    }
    __syncthreads();

    const int wave = tid >> 5;
    const int lane = tid & 31;
    const int m    = lane & 15;    // M within tile (A) / N within tile (B,C,D)
    const int h    = lane >> 4;    // lane half

    const int q = qblock + wave * 16 + m;          // A-matrix row fed by this lane
    const bool qvalid = (q < Q_DIM);
    const float* __restrict__ arow =
        logits + ((size_t)b * Q_DIM + (qvalid ? q : 0)) * T_DIM;

    // per-lane base into the fragment-ready B
    const _Float16* __restrict__ bb = ldsB + lane * 16;

    v8f acc[6] = {};  // 6 N-tiles of 16 -> 96 columns

    #pragma unroll 2
    for (int k0 = 0; k0 < T_DIM; k0 += 32) {
        if (k0 + 32 < T_DIM)
            __builtin_prefetch(arow + k0 + 32, 0, 0);   // global_prefetch_b8

        // --- A fragment: contiguous k order used here (same as ldsB):
        //   element j -> K = k0 + 16*h + j
        //   i.e. one contiguous 16-float chunk at k0+16h.
        float av[16];
        if (qvalid) {
            const float* pa = arow + k0 + 16 * h;
            const v4f x0 = *(const v4f*)(pa);
            const v4f x1 = *(const v4f*)(pa + 4);
            const v4f x2 = *(const v4f*)(pa + 8);
            const v4f x3 = *(const v4f*)(pa + 12);
            #pragma unroll
            for (int j = 0; j < 4; ++j) {
                av[j]      = x0[j];
                av[4 + j]  = x1[j];
                av[8 + j]  = x2[j];
                av[12 + j] = x3[j];
            }
        } else {
            #pragma unroll
            for (int j = 0; j < 16; ++j) av[j] = 0.0f;
        }
        v16h afrag;
        #pragma unroll
        for (int j = 0; j < 16; ++j)
            afrag[j] = (_Float16)sigmoidf_fast(av[j]);

        // --- Load all 6 B fragments first (DS clause), then 6 WMMAs.
        const _Float16* __restrict__ bk = bb + (k0 >> 5) * (NTILES * 32 * 16);
        v16h bfrag[6];
        #pragma unroll
        for (int n = 0; n < 6; ++n)
            bfrag[n] = *(const v16h*)(bk + n * (32 * 16));
        #pragma unroll
        for (int n = 0; n < 6; ++n)
            acc[n] = __builtin_amdgcn_wmma_f32_16x16x32_f16(
                false, afrag, false, bfrag[n],
                (short)0, acc[n], false, false);
    }

    // --- Store C/D: VGPR r -> M = r + 8*h, N = lane%16.
    #pragma unroll
    for (int n = 0; n < 6; ++n) {
        #pragma unroll
        for (int r = 0; r < 8; ++r) {
            const int qq = qblock + wave * 16 + r + 8 * h;
            if (qq < Q_DIM)
                prob[((size_t)b * Q_DIM + qq) * C_PAD + n * 16 + m] = acc[n][r];
        }
    }
}

// ---------------------------------------------------------------------------
// Kernel 2: per-batch exact top-300 via 4-round 8-bit radix select on the
// float bit pattern (all probs >= 0, so uint order == float order), then
// candidate collection + bitonic sort (descending) + box gather/scale.
// One 1024-thread block per batch.
// ---------------------------------------------------------------------------
__global__ __launch_bounds__(TOPK_THREADS)
void topk_kernel(const float* __restrict__ prob,
                 const float* __restrict__ pred_boxes,
                 const float* __restrict__ tsizes,
                 float* __restrict__ out_scores,
                 float* __restrict__ out_labels,
                 float* __restrict__ out_boxes)
{
    __shared__ unsigned hist[256];
    __shared__ unsigned s_prefix, s_k, s_ncand;
    __shared__ float cval[CAND_CAP];
    __shared__ int   cidx[CAND_CAP];

    const int b   = blockIdx.x;
    const int tid = threadIdx.x;
    const float* __restrict__ p = prob + (size_t)b * Q_DIM * C_PAD;
    const int NTOT = Q_DIM * C_PAD;

    if (tid == 0) { s_prefix = 0u; s_k = NUM_SEL; s_ncand = 0u; }

    // ---- 4 radix-select rounds, high byte -> low byte.
    for (int shift = 24; shift >= 0; shift -= 8) {
        for (int i = tid; i < 256; i += TOPK_THREADS) hist[i] = 0u;
        __syncthreads();
        const unsigned prefix = s_prefix;
        const unsigned pmask  = (shift == 24) ? 0u : (0xFFFFFFFFu << (shift + 8));
        for (int i = tid; i < NTOT; i += TOPK_THREADS) {
            const int c = i % C_PAD;
            if (c >= C_DIM) continue;          // skip padded columns
            const unsigned key = __float_as_uint(p[i]);
            if ((key & pmask) == prefix)
                atomicAdd(&hist[(key >> shift) & 0xFFu], 1u);
        }
        __syncthreads();
        if (tid == 0) {
            unsigned k = s_k, run = 0u;
            int d = 255;
            for (; d >= 0; --d) {
                const unsigned cnt = hist[d];
                if (run + cnt >= k) { s_k = k - run; break; }
                run += cnt;
            }
            if (d < 0) d = 0;
            s_prefix = prefix | ((unsigned)d << shift);
        }
        __syncthreads();
    }

    // ---- Collect candidates with key >= exact 300th-largest key.
    const unsigned kth = s_prefix;
    for (int i = tid; i < NTOT; i += TOPK_THREADS) {
        const int c = i % C_PAD;
        if (c >= C_DIM) continue;
        const float v = p[i];
        if (__float_as_uint(v) >= kth) {
            const unsigned pos = atomicAdd(&s_ncand, 1u);
            if (pos < CAND_CAP) { cval[pos] = v; cidx[pos] = i; }
        }
    }
    __syncthreads();
    const int ncand = (s_ncand < CAND_CAP) ? (int)s_ncand : CAND_CAP;
    for (int i = tid; i < CAND_CAP; i += TOPK_THREADS) {
        if (i >= ncand) { cval[i] = -__builtin_inff(); cidx[i] = -1; }
    }

    // ---- Bitonic sort, descending. 2048 elements, 1024 compare pairs/substage.
    for (unsigned size = 2; size <= CAND_CAP; size <<= 1) {
        for (unsigned stride = size >> 1; stride > 0; stride >>= 1) {
            __syncthreads();
            const unsigned t = (unsigned)tid;
            const unsigned i = 2u * t - (t & (stride - 1u));
            const unsigned j = i + stride;
            const bool desc = ((i & size) == 0u);
            const float vi = cval[i], vj = cval[j];
            if ((vi < vj) == desc) {
                cval[i] = vj; cval[j] = vi;
                const int ti = cidx[i]; cidx[i] = cidx[j]; cidx[j] = ti;
            }
        }
    }
    __syncthreads();

    // ---- Emit scores / labels / scaled xyxy boxes.
    const float img_h = tsizes[b * 2 + 0];
    const float img_w = tsizes[b * 2 + 1];
    for (int t = tid; t < NUM_SEL; t += TOPK_THREADS) {
        const float v  = cval[t];
        const int  idx = cidx[t];
        const int  qq  = (idx >= 0) ? (idx / C_PAD) : 0;
        const int  cc  = (idx >= 0) ? (idx - qq * C_PAD) : 0;

        out_scores[b * NUM_SEL + t] = v;
        out_labels[b * NUM_SEL + t] = (float)cc;

        const float* bx = pred_boxes + ((size_t)b * Q_DIM + qq) * 4;
        const float cx = bx[0], cy = bx[1], w = bx[2], hh = bx[3];
        float* ob = out_boxes + ((size_t)b * NUM_SEL + t) * 4;
        ob[0] = (cx - 0.5f * w)  * img_w;
        ob[1] = (cy - 0.5f * hh) * img_h;
        ob[2] = (cx + 0.5f * w)  * img_w;
        ob[3] = (cy + 0.5f * hh) * img_h;
    }
}

// ---------------------------------------------------------------------------
extern "C" void kernel_launch(void* const* d_in, const int* in_sizes, int n_in,
                              void* d_out, int out_size, void* d_ws, size_t ws_size,
                              hipStream_t stream)
{
    const float* pred_logits  = (const float*)d_in[0]; // (128, 900, 256)
    const float* pred_boxes   = (const float*)d_in[1]; // (128, 900, 4)
    const float* positive_map = (const float*)d_in[2]; // (91, 256)
    const float* target_sizes = (const float*)d_in[3]; // (128, 2)

    float* out        = (float*)d_out;
    float* out_scores = out;                              // 128*300
    float* out_labels = out + B_BATCH * NUM_SEL;          // 128*300
    float* out_boxes  = out + 2 * B_BATCH * NUM_SEL;      // 128*300*4

    float* prob = (float*)d_ws;   // 128*900*96 f32 = 44.2 MB scratch

    dim3 g1((Q_DIM + 63) / 64, B_BATCH);   // (15, 128)
    sigmoid_gemm_wmma<<<g1, 128, 0, stream>>>(pred_logits, positive_map, prob);

    topk_kernel<<<B_BATCH, TOPK_THREADS, 0, stream>>>(
        prob, pred_boxes, target_sizes, out_scores, out_labels, out_boxes);
}
